// FeatureWeightNet_86955907875357
// MI455X (gfx1250) — matrix operand order repaired and checked by reading, not compiled, not run on hardware
//
#include <hip/hip_runtime.h>
#include <hip/hip_bf16.h>
#include <math.h>

// CDNA5 (gfx1250) wave32 WMMA types
typedef __attribute__((ext_vector_type(16))) _Float16 v16h;
typedef __attribute__((ext_vector_type(8)))  _Float16 v8h;
typedef __attribute__((ext_vector_type(8)))  float    v8f;
typedef __attribute__((ext_vector_type(2)))  float    v2f;

#define G_        8
#define NB_       9
#define H_        256
#define W_        320
#define C_        64
#define HW_       (H_ * W_)
#define BN_EPS_   1e-5f

// Workspace layout (bytes):
//   [0,   256)  w0h : 16x8  f16  BN0-scaled conv0 weights
//   [256, 512)  w1h : 8x16  f16  BN1-scaled conv1 weights
//   [512, 576)  b0f : 16    f32  folded BN0 bias
//   [576, 608)  b1f : 8     f32  folded BN1 bias
#define WS_W0_   0
#define WS_W1_   256
#define WS_B0_   512
#define WS_B1_   576

struct __align__(4) f2u { float x, y; };

// ---------- tiny fold kernel: BN -> scaled f16 weights + f32 bias ----------
__global__ void fwn_fold_kernel(const float* __restrict__ conv0_w,
                                const float* __restrict__ bn0_g, const float* __restrict__ bn0_b,
                                const float* __restrict__ bn0_m, const float* __restrict__ bn0_v,
                                const float* __restrict__ conv1_w,
                                const float* __restrict__ bn1_g, const float* __restrict__ bn1_b,
                                const float* __restrict__ bn1_m, const float* __restrict__ bn1_v,
                                void* __restrict__ ws)
{
    _Float16* w0h = (_Float16*)((char*)ws + WS_W0_);
    _Float16* w1h = (_Float16*)((char*)ws + WS_W1_);
    float*    b0f = (float*)((char*)ws + WS_B0_);
    float*    b1f = (float*)((char*)ws + WS_B1_);

    const int t = threadIdx.x;
    if (t < 16) {
        const float s = bn0_g[t] * rsqrtf(bn0_v[t] + BN_EPS_);
        #pragma unroll
        for (int j = 0; j < 8; ++j)
            w0h[t * 8 + j] = (_Float16)(s * conv0_w[t * 8 + j]);
        b0f[t] = bn0_b[t] - bn0_m[t] * s;
    } else if (t < 24) {
        const int r = t - 16;
        const float s = bn1_g[r] * rsqrtf(bn1_v[r] + BN_EPS_);
        #pragma unroll
        for (int j = 0; j < 16; ++j)
            w1h[r * 16 + j] = (_Float16)(s * conv1_w[r * 16 + j]);
        b1f[r] = bn1_b[r] - bn1_m[r] * s;
    }
}

// ---------- fused main kernel ----------
// One thread = one output pixel for the gather stage (L2-gather bound part).
// Each wave32 runs the per-pixel MLP as two 16-pixel WMMA tiles per layer,
// with fragment layouts chosen so NO cross-lane data movement is needed.
__global__ __launch_bounds__(256)
void fwn_fused_kernel(const float* __restrict__ ref_feature,
                      const float* __restrict__ grid,
                      const void*  __restrict__ ws,
                      const float* __restrict__ sim_w, const float* __restrict__ sim_b,
                      float* __restrict__ out, int total)
{
    const int tid  = blockIdx.x * blockDim.x + threadIdx.x;
    const int lane = threadIdx.x & 31;
    const int lrow = lane & 15;
    const bool hi  = lane >= 16;

    // Clamp so every lane does valid work (keeps EXEC all-ones for WMMA).
    const int p = (tid < total) ? tid : (total - 1);

    // ---- decompose pixel index: [B, NB, H, W] ----
    const int w  = p % W_;
    const int h  = (p / W_) % H_;
    const int n  = (p / HW_) % NB_;
    const int b  = p / (HW_ * NB_);

    // ---------------- Stage 1: grid_sample(border) + group correlation ----------------
    const v2f* gp = reinterpret_cast<const v2f*>(grid);
    const v2f g2  = __builtin_nontemporal_load(&gp[((b * NB_ + n) * H_ + h) * W_ + w]);

    float ix = ((g2[0] + 1.0f) * (float)W_ - 1.0f) * 0.5f;
    float iy = ((g2[1] + 1.0f) * (float)H_ - 1.0f) * 0.5f;
    ix = fminf(fmaxf(ix, 0.0f), (float)(W_ - 1));
    iy = fminf(fmaxf(iy, 0.0f), (float)(H_ - 1));
    const float x0f = floorf(ix), y0f = floorf(iy);
    const float fx = ix - x0f, fy = iy - y0f;
    int x0i = (int)x0f; x0i = min(max(x0i, 0), W_ - 1);
    int y0i = (int)y0f; y0i = min(max(y0i, 0), H_ - 1);

    // Fold the border clamp into the 2x2 weights on a base cell clamped in-range:
    // if x0==W-1 then fx==0 and weights (0,1) on columns (W-2,W-1) give the border value.
    const int xa = min(x0i, W_ - 2);
    const int ya = min(y0i, H_ - 2);
    const bool xb = (x0i == W_ - 1);
    const bool yb = (y0i == H_ - 1);
    const float wx0 = xb ? 0.0f : (1.0f - fx);
    const float wx1 = xb ? 1.0f : fx;
    const float wy0 = yb ? 0.0f : (1.0f - fy);
    const float wy1 = yb ? 1.0f : fy;
    const float wA = wx0 * wy0, wB = wx1 * wy0;
    const float wC = wx0 * wy1, wD = wx1 * wy1;

    const int o0   = ya * W_ + xa;       // top row pair
    const int o1   = o0 + W_;            // bottom row pair
    const int oref = h * W_ + w;
    const float* featB = ref_feature + (size_t)b * C_ * HW_;

    float corr[G_];
    #pragma unroll
    for (int g = 0; g < G_; ++g) corr[g] = 0.0f;

    #pragma unroll 8
    for (int c = 0; c < C_; ++c) {
        const float* pl = featB + c * HW_;
        const f2u r0 = *reinterpret_cast<const f2u*>(pl + o0);
        const f2u r1 = *reinterpret_cast<const f2u*>(pl + o1);
        const float v = wA * r0.x + wB * r0.y + wC * r1.x + wD * r1.y;
        corr[c >> 3] = fmaf(v, pl[oref], corr[c >> 3]);
    }

    // corr in f16 (once, reused by both tiles); includes the mean over Cg=8.
    _Float16 ch[8];
    #pragma unroll
    for (int g = 0; g < G_; ++g) ch[g] = (_Float16)(corr[g] * 0.125f);

    // ---------------- Stage 2: MLP via v_wmma_f32_16x16x32_f16 ----------------
    const _Float16* w0h = (const _Float16*)((const char*)ws + WS_W0_);
    const _Float16* w1h = (const _Float16*)((const char*)ws + WS_W1_);
    const float*    b0f = (const float*)((const char*)ws + WS_B0_);
    const float*    b1f = (const float*)((const char*)ws + WS_B1_);

    // A0  : lanes 0-15 hold row=lrow weights at halfs 0..7 (lane half 0)  -> tile 0
    // A0s : lanes 16-31 hold row=lrow weights at halfs 0..7 (lane half 1) -> tile 1
    const v8h w0row = *reinterpret_cast<const v8h*>(w0h + lrow * 8);
    v16h a0, a0s;
    #pragma unroll
    for (int i = 0; i < 16; ++i) { a0[i] = (_Float16)0.0f; a0s[i] = (_Float16)0.0f; }
    #pragma unroll
    for (int j = 0; j < 8; ++j) {
        const _Float16 wv = hi ? (_Float16)0.0f : w0row[j];
        a0[j]      = wv;
        a0s[j]     = hi ? w0row[j] : (_Float16)0.0f;
    }
    // C0: bias0[r] for lanes 0-15 (rows 0-7), bias0[r+8] for lanes 16-31 (rows 8-15).
    const v8f c0 = *reinterpret_cast<const v8f*>(b0f + (hi ? 8 : 0));

    // A1 (8x16 weights, rows 8-15 zero): lanes 0-15 hold input ch 0..7 at halfs 0..7
    // and lanes 16-31 hold input ch 8..15 at halfs 0..7; halfs 8..15 zero. This matches a
    // B fragment where every lane contributes its OWN d0 values.
    const v16h w1row = *reinterpret_cast<const v16h*>(w1h + (lrow & 7) * 16);
    const bool rowok = (!hi) && (lrow < 8);
    v16h a1;
    #pragma unroll
    for (int j = 0; j < 16; ++j) a1[j] = (lrow < 8 && j < 8) ? w1row[(hi ? 8 : 0) + j] : (_Float16)0.0f;

    const v8f c1raw = *reinterpret_cast<const v8f*>(b1f);
    v8f c1;
    #pragma unroll
    for (int r = 0; r < 8; ++r) c1[r] = hi ? 0.0f : c1raw[r];  // rows 8-15 don't-care

    float sw[8];
    #pragma unroll
    for (int r = 0; r < 8; ++r) sw[r] = sim_w[r];
    const float sb = sim_b[0];

    const int waveBase = tid & ~31;

    // =================== Tile 0: pixels waveBase + 0..15 ===================
    {
        v16h b0;   // lo lane l = column l, K=0..7 = its own corr; hi lanes zero.
        #pragma unroll
        for (int i = 0; i < 16; ++i) b0[i] = (_Float16)0.0f;
        #pragma unroll
        for (int k = 0; k < 8; ++k) b0[k] = hi ? (_Float16)0.0f : ch[k];

        v8f d0 = __builtin_amdgcn_wmma_f32_16x16x32_f16(
            false, a0, false, b0, (short)0, c0, false, false);

        v16h b1;   // every lane contributes its own relu(d0) — no shuffles.
        #pragma unroll
        for (int i = 8; i < 16; ++i) b1[i] = (_Float16)0.0f;
        #pragma unroll
        for (int j = 0; j < 8; ++j) b1[j] = (_Float16)fmaxf(d0[j], 0.0f);

        v8f d1 = __builtin_amdgcn_wmma_f32_16x16x32_f16(
            false, a1, false, b1, (short)0, c1, false, false);

        float s = sb;
        #pragma unroll
        for (int r = 0; r < 8; ++r) s = fmaf(sw[r], fmaxf(d1[r], 0.0f), s);
        const float o = 1.0f / (1.0f + __expf(-s));

        const int outIdx = waveBase + lrow;
        if (!hi && outIdx < total) __builtin_nontemporal_store(o, &out[outIdx]);
    }

    // =================== Tile 1: pixels waveBase + 16..31 ===================
    {
        v16h b0;   // hi lane 16+l = column l, K=16..23 = its own corr; lo lanes zero.
        #pragma unroll
        for (int i = 0; i < 16; ++i) b0[i] = (_Float16)0.0f;
        #pragma unroll
        for (int k = 0; k < 8; ++k) b0[k] = hi ? ch[k] : (_Float16)0.0f;

        v8f d0 = __builtin_amdgcn_wmma_f32_16x16x32_f16(
            false, a0s, false, b0, (short)0, c0, false, false);

        v16h b1;
        #pragma unroll
        for (int i = 8; i < 16; ++i) b1[i] = (_Float16)0.0f;
        #pragma unroll
        for (int j = 0; j < 8; ++j) b1[j] = (_Float16)fmaxf(d0[j], 0.0f);

        v8f d1 = __builtin_amdgcn_wmma_f32_16x16x32_f16(
            false, a1, false, b1, (short)0, c1, false, false);

        float s = sb;
        #pragma unroll
        for (int r = 0; r < 8; ++r) s = fmaf(sw[r], fmaxf(d1[r], 0.0f), s);
        const float o = 1.0f / (1.0f + __expf(-s));

        const int outIdx = waveBase + 16 + lrow;
        if (!hi && outIdx < total) __builtin_nontemporal_store(o, &out[outIdx]);
    }
}

extern "C" void kernel_launch(void* const* d_in, const int* in_sizes, int n_in,
                              void* d_out, int out_size, void* d_ws, size_t ws_size,
                              hipStream_t stream) {
    const float* ref_feature = (const float*)d_in[0];
    const float* grid        = (const float*)d_in[1];
    const float* conv0_w     = (const float*)d_in[2];
    const float* bn0_g       = (const float*)d_in[3];
    const float* bn0_b       = (const float*)d_in[4];
    const float* bn0_m       = (const float*)d_in[5];
    const float* bn0_v       = (const float*)d_in[6];
    const float* conv1_w     = (const float*)d_in[7];
    const float* bn1_g       = (const float*)d_in[8];
    const float* bn1_b       = (const float*)d_in[9];
    const float* bn1_m       = (const float*)d_in[10];
    const float* bn1_v       = (const float*)d_in[11];
    const float* sim_w       = (const float*)d_in[12];
    const float* sim_b       = (const float*)d_in[13];
    float* out = (float*)d_out;

    const int total = in_sizes[1] / 2;            // B*NB*H*W grid points
    const int blocks = (total + 255) / 256;

    fwn_fold_kernel<<<1, 32, 0, stream>>>(
        conv0_w, bn0_g, bn0_b, bn0_m, bn0_v,
        conv1_w, bn1_g, bn1_b, bn1_m, bn1_v, d_ws);

    fwn_fused_kernel<<<blocks, 256, 0, stream>>>(
        ref_feature, grid, d_ws, sim_w, sim_b, out, total);
}
